// Vectorizer_5085241279191
// MI455X (gfx1250) — matrix operand order repaired and checked by reading, not compiled, not run on hardware
//
#include <hip/hip_runtime.h>
#include <hip/hip_bf16.h>
#include <math.h>

typedef __attribute__((ext_vector_type(16))) __bf16 v16bf;
typedef __attribute__((ext_vector_type(8)))  float  v8f;

#define HDIM   768
#define NHEAD  12
#define DHEAD  64
#define FFDIM  3072
#define NLAYER 6
#define WINSZ  256
#define NSPAN  32
#define PDIM   256

__device__ inline float gelu_exact(float x) {
  return 0.5f * x * (1.0f + erff(x * 0.7071067811865476f));
}
__device__ inline v8f zero8() {
  v8f z;
#pragma unroll
  for (int i = 0; i < 8; ++i) z[i] = 0.0f;
  return z;
}

// CDNA5 async global->LDS copy (ASYNCcnt-tracked, no VGPR staging).
// VDST carries the wave-relative LDS byte offset (= low 32 bits of generic LDS pointer).
__device__ inline void async_ld_b128(void* lds, const void* g) {
  uint32_t loff = (uint32_t)(uintptr_t)lds;
  asm volatile("global_load_async_to_lds_b128 %0, %1, off" :: "v"(loff), "v"(g) : "memory");
}
__device__ inline void wait_async0() {
  asm volatile("s_wait_asynccnt 0" ::: "memory");
}

// ---------------------------------------------------------------- weight pack: fp32 [K][N] -> bf16 WMMA-fragment-major
// Packed element for (k,n): tile (kt=k>>5, nt=n>>4); lane = (n&15) | (((k>>4)&1)<<4); elem = k&15
// offset = ((kt*(N/16)+nt)*32 + lane)*16 + elem   -> a 32x64 staging tile is 4KB contiguous,
// and each lane's fragment is one contiguous 32B LDS read.
__global__ void k_pack_w(const float* __restrict__ W, __bf16* __restrict__ P,
                         long layers, int K, int N) {
  long per = (long)K * N;
  long total = layers * per;
  long stride = (long)gridDim.x * blockDim.x;
  for (long idx = (long)blockIdx.x * blockDim.x + threadIdx.x; idx < total; idx += stride) {
    long l = idx / per;
    long r = idx - l * per;
    int k = (int)(r / N), n = (int)(r % N);
    int lane = (n & 15) | (((k >> 4) & 1) << 4);
    long dst = l * per + (((long)(k >> 5) * (N >> 4) + (n >> 4)) * 32 + lane) * 16 + (k & 15);
    P[dst] = (__bf16)W[idx];
  }
}

// ---------------------------------------------------------------- embedding
__global__ void k_embed(const int* __restrict__ ids, const float* __restrict__ wemb,
                        const float* __restrict__ pemb, float* __restrict__ x, int S) {
  int row = blockIdx.x;
  int tok = ids[row];
  int spos = row % S;
  for (int i = threadIdx.x; i < HDIM; i += blockDim.x)
    x[(long)row * HDIM + i] = wemb[(long)tok * HDIM + i] + pemb[(long)spos * HDIM + i];
}

// ---------------------------------------------------------------- layernorm (+optional residual, +optional bf16 copy)
__global__ __launch_bounds__(256) void k_ln(const float* __restrict__ x, const float* __restrict__ res,
                                            const float* __restrict__ g, const float* __restrict__ b,
                                            float* __restrict__ out, __bf16* __restrict__ outbf) {
  __shared__ float red[256];
  int row = blockIdx.x, tid = threadIdx.x;
  const float* xr = x + (long)row * HDIM;
  const float* rr = res ? res + (long)row * HDIM : nullptr;
  float s = 0.f;
  for (int i = tid; i < HDIM; i += 256) { float v = xr[i] + (rr ? rr[i] : 0.f); s += v; }
  red[tid] = s; __syncthreads();
  for (int o = 128; o > 0; o >>= 1) { if (tid < o) red[tid] += red[tid + o]; __syncthreads(); }
  float mean = red[0] / HDIM; __syncthreads();
  s = 0.f;
  for (int i = tid; i < HDIM; i += 256) { float v = xr[i] + (rr ? rr[i] : 0.f) - mean; s += v * v; }
  red[tid] = s; __syncthreads();
  for (int o = 128; o > 0; o >>= 1) { if (tid < o) red[tid] += red[tid + o]; __syncthreads(); }
  float rstd = rsqrtf(red[0] / HDIM + 1e-5f);
  for (int i = tid; i < HDIM; i += 256) {
    float v = (xr[i] + (rr ? rr[i] : 0.f) - mean) * rstd * g[i] + b[i];
    out[(long)row * HDIM + i] = v;
    if (outbf) outbf[(long)row * HDIM + i] = (__bf16)v;
  }
}

// ---------------------------------------------------------------- tiled WMMA GEMM, double-buffered async global->LDS
// C = act(A*Bpacked + bias), optional bf16 copy of C.
// A: row-major bf16 [M][K]; Bp: fragment-major packed bf16 (see k_pack_w).
__global__ __launch_bounds__(256) void k_gemm_bf16(
    const __bf16* __restrict__ A, const __bf16* __restrict__ Bp,
    const float* __restrict__ bias, float* __restrict__ C, __bf16* __restrict__ Cbf,
    int M, int N, int K, int act) {
  __shared__ alignas(32) __bf16 As[2][128 * 32]; // swizzled: ((mg*32+lane)*16 + 8*run)
  __shared__ alignas(32) __bf16 Bs[2][32 * 64];  // packed subtiles, contiguous
  int tid = threadIdx.x;
  int lane = tid & 31, wid = tid >> 5;
  int wm = (wid & 3) * 32;                       // 4 waves along M
  int wn = (wid >> 2) * 32;                      // 2 waves along N
  long m0 = (long)blockIdx.y * 128;
  long n0 = (long)blockIdx.x * 64;
  int nt16 = N >> 4;

  // A tile: 512 x 16B chunks; chunk c: run=c&1, h=(c>>1)&1, ll=(c>>2)&15, mg=c>>6
  auto issueA = [&](int k0, int buf) {
#pragma unroll
    for (int c2 = 0; c2 < 2; ++c2) {
      int c = tid + (c2 << 8);
      int run = c & 1, h = (c >> 1) & 1, ll = (c >> 2) & 15, mg = c >> 6;
      const __bf16* src = A + (m0 + mg * 16 + ll) * (long)K + k0 + 16 * h + 8 * run;
      __bf16* dst = &As[buf][((mg * 32) + ll + 16 * h) * 16 + 8 * run];
      async_ld_b128(dst, src);
    }
  };
  // B tile: 2048 halfs contiguous in packed layout
  auto issueB = [&](int k0, int buf) {
    long tileBase = ((long)(k0 >> 5) * nt16 + (n0 >> 4)) * 512;
    async_ld_b128(&Bs[buf][tid * 8], Bp + tileBase + tid * 8);
  };

  v8f acc[2][2];
#pragma unroll
  for (int i = 0; i < 2; ++i)
#pragma unroll
    for (int j = 0; j < 2; ++j) acc[i][j] = zero8();

  int nk = K >> 5;
  issueA(0, 0);
  issueB(0, 0);
  for (int ki = 0; ki < nk; ++ki) {
    int buf = ki & 1;
    wait_async0();            // own-wave async data landed in LDS
    __syncthreads();          // ...and every other wave's too
    if (ki + 1 < nk) { issueA((ki + 1) << 5, buf ^ 1); issueB((ki + 1) << 5, buf ^ 1); }
    v16bf a0 = *(const v16bf*)&As[buf][(((wm >> 4) + 0) * 32 + lane) * 16];
    v16bf a1 = *(const v16bf*)&As[buf][(((wm >> 4) + 1) * 32 + lane) * 16];
    v16bf b0 = *(const v16bf*)&Bs[buf][(((wn >> 4) + 0) * 32 + lane) * 16];
    v16bf b1 = *(const v16bf*)&Bs[buf][(((wn >> 4) + 1) * 32 + lane) * 16];
    acc[0][0] = __builtin_amdgcn_wmma_f32_16x16x32_bf16(false, a0, false, b0, (short)0, acc[0][0], false, false);
    acc[0][1] = __builtin_amdgcn_wmma_f32_16x16x32_bf16(false, a0, false, b1, (short)0, acc[0][1], false, false);
    acc[1][0] = __builtin_amdgcn_wmma_f32_16x16x32_bf16(false, a1, false, b0, (short)0, acc[1][0], false, false);
    acc[1][1] = __builtin_amdgcn_wmma_f32_16x16x32_bf16(false, a1, false, b1, (short)0, acc[1][1], false, false);
  }
  int rl = (lane >> 4) << 3, cn = lane & 15;
#pragma unroll
  for (int tm = 0; tm < 2; ++tm)
#pragma unroll
    for (int tn = 0; tn < 2; ++tn)
#pragma unroll
      for (int r = 0; r < 8; ++r) {
        long row = m0 + wm + 16 * tm + r + rl;
        long col = n0 + wn + 16 * tn + cn;
        float v = acc[tm][tn][r] + bias[col];
        if (act) v = gelu_exact(v);
        C[row * N + col] = v;
        if (Cbf) Cbf[row * N + col] = (__bf16)v;
      }
}

// ---------------------------------------------------------------- WMMA sliding-window attention
// One wave per (b,h,chunk,16-q-row tile). Scores 16x768 via WMMA, softmax, P*V via WMMA.
__global__ __launch_bounds__(32) void k_attn(
    const float* __restrict__ q, const float* __restrict__ k, const float* __restrict__ v,
    const int* __restrict__ amask, float* __restrict__ out, __bf16* __restrict__ outbf, int S) {
  __shared__ __bf16 qs[16 * 64];
  __shared__ __bf16 ks[16 * 64];
  __shared__ __bf16 vs[32 * 64];
  __shared__ float sc[16 * 768];
  int lane = threadIdx.x;
  int qt = blockIdx.x;                // 0..15 (16 q rows each)
  int c  = blockIdx.y;                // chunk
  int bh = blockIdx.z;
  int b = bh / NHEAD, h = bh % NHEAD;
  long tokBase = (long)b * S + (long)c * WINSZ + qt * 16;
  int colBase = h * DHEAD;

  for (int i = lane; i < 16 * 64; i += 32) {
    int r = i >> 6, d = i & 63;
    qs[i] = (__bf16)q[(tokBase + r) * HDIM + colBase + d];
  }
  __syncthreads();

  // ---- scores = Q * K^T over 768-key window
  for (int kt = 0; kt < 48; ++kt) {
    for (int i = lane; i < 16 * 64; i += 32) {
      int r = i >> 6, d = i & 63;
      long kp = (long)c * WINSZ - WINSZ + kt * 16 + r;
      float val = (kp >= 0 && kp < S) ? k[((long)b * S + kp) * HDIM + colBase + d] : 0.f;
      ks[i] = (__bf16)val;
    }
    __syncthreads();
    v16bf a0, a1, b0, b1;
    {
      int m = lane & 15; int khalf = (lane >> 4) << 4;
#pragma unroll
      for (int vv = 0; vv < 8; ++vv) {
        int kk = (vv << 1) + khalf;
        a0[2 * vv] = qs[m * 64 + kk];      a0[2 * vv + 1] = qs[m * 64 + kk + 1];
        a1[2 * vv] = qs[m * 64 + 32 + kk]; a1[2 * vv + 1] = qs[m * 64 + 32 + kk + 1];
      }
      int n = lane & 15; int kb = (lane >> 4) << 4;
#pragma unroll
      for (int vv = 0; vv < 8; ++vv) {
        int kk = kb + (vv << 1);
        b0[2 * vv] = ks[n * 64 + kk];      b0[2 * vv + 1] = ks[n * 64 + kk + 1];
        b1[2 * vv] = ks[n * 64 + 32 + kk]; b1[2 * vv + 1] = ks[n * 64 + 32 + kk + 1];
      }
    }
    v8f s = zero8();
    s = __builtin_amdgcn_wmma_f32_16x16x32_bf16(false, a0, false, b0, (short)0, s, false, false);
    s = __builtin_amdgcn_wmma_f32_16x16x32_bf16(false, a1, false, b1, (short)0, s, false, false);
    int rl = (lane >> 4) << 3;
    int jn = kt * 16 + (lane & 15);
    long kp = (long)c * WINSZ - WINSZ + jn;
    bool kvalid = (kp >= 0 && kp < S) && (amask[(long)b * S + (kp < 0 ? 0 : (kp >= S ? S - 1 : kp))] > 0);
#pragma unroll
    for (int r = 0; r < 8; ++r) {
      int rowL = r + rl;
      int iq = qt * 16 + rowL;
      bool band = (jn >= iq) && (jn <= iq + 2 * WINSZ);
      sc[rowL * 768 + jn] = (band && kvalid) ? s[r] * 0.125f : -1.0e9f;
    }
    __syncthreads();
  }

  // ---- softmax per q row (wave-cooperative)
  for (int r = 0; r < 16; ++r) {
    float mx = -3.0e38f;
    for (int j = lane; j < 768; j += 32) mx = fmaxf(mx, sc[r * 768 + j]);
#pragma unroll
    for (int o = 16; o > 0; o >>= 1) mx = fmaxf(mx, __shfl_xor(mx, o, 32));
    float sum = 0.f;
    for (int j = lane; j < 768; j += 32) { float e = __expf(sc[r * 768 + j] - mx); sc[r * 768 + j] = e; sum += e; }
#pragma unroll
    for (int o = 16; o > 0; o >>= 1) sum += __shfl_xor(sum, o, 32);
    float inv = 1.f / sum;
    for (int j = lane; j < 768; j += 32) sc[r * 768 + j] *= inv;
  }
  __syncthreads();

  // ---- out = P * V
  v8f oacc[4];
#pragma unroll
  for (int i = 0; i < 4; ++i) oacc[i] = zero8();
  for (int kt = 0; kt < 24; ++kt) {
    for (int i = lane; i < 32 * 64; i += 32) {
      int r = i >> 6, d = i & 63;
      long kp = (long)c * WINSZ - WINSZ + kt * 32 + r;
      vs[i] = (__bf16)((kp >= 0 && kp < S) ? v[((long)b * S + kp) * HDIM + colBase + d] : 0.f);
    }
    __syncthreads();
    v16bf pa;
    {
      int m = lane & 15; int khalf = (lane >> 4) << 4;
#pragma unroll
      for (int vv = 0; vv < 8; ++vv) {
        int kk = (vv << 1) + khalf;
        pa[2 * vv]     = (__bf16)sc[m * 768 + kt * 32 + kk];
        pa[2 * vv + 1] = (__bf16)sc[m * 768 + kt * 32 + kk + 1];
      }
    }
#pragma unroll
    for (int n = 0; n < 4; ++n) {
      v16bf vb;
      int nn = n * 16 + (lane & 15);
      int kb = (lane >> 4) << 4;
#pragma unroll
      for (int vv = 0; vv < 8; ++vv) {
        int kk = kb + (vv << 1);
        vb[2 * vv] = vs[kk * 64 + nn]; vb[2 * vv + 1] = vs[(kk + 1) * 64 + nn];
      }
      oacc[n] = __builtin_amdgcn_wmma_f32_16x16x32_bf16(false, pa, false, vb, (short)0, oacc[n], false, false);
    }
    __syncthreads();
  }
  int rl = (lane >> 4) << 3, cn = lane & 15;
#pragma unroll
  for (int n = 0; n < 4; ++n)
#pragma unroll
    for (int r = 0; r < 8; ++r) {
      long tok = tokBase + r + rl;
      int col = colBase + n * 16 + cn;
      float val = oacc[n][r];
      out[tok * HDIM + col] = val;
      outbf[tok * HDIM + col] = (__bf16)val;
    }
}

// ---------------------------------------------------------------- span pooling
__global__ __launch_bounds__(256) void k_span_pool(
    const float* __restrict__ enc, const int* __restrict__ span_mask,
    const int* __restrict__ sidx, const float* __restrict__ pw,
    float* __restrict__ emb, int S) {
  __shared__ float sscore[2048];
  __shared__ float red[256];
  int n = blockIdx.x, tid = threadIdx.x;
  const float* seq = enc + (long)sidx[n] * S * HDIM;
  for (int t = tid; t < S; t += 256) {
    float d = 0.f;
    const float* rowp = seq + (long)t * HDIM;
    for (int cc = 0; cc < HDIM; ++cc) d += rowp[cc] * pw[cc];
    d = fminf(fmaxf(d, -100.f), 100.f);
    float am = (span_mask[(long)n * S + t] == 1) ? 1.f : 0.f;
    sscore[t] = d * am + (-1.0e9f) * (1.f - am);
  }
  __syncthreads();
  float mx = -3.0e38f;
  for (int t = tid; t < S; t += 256) mx = fmaxf(mx, sscore[t]);
  red[tid] = mx; __syncthreads();
  for (int o = 128; o > 0; o >>= 1) { if (tid < o) red[tid] = fmaxf(red[tid], red[tid + o]); __syncthreads(); }
  mx = red[0]; __syncthreads();
  float sum = 0.f;
  for (int t = tid; t < S; t += 256) { float e = __expf(sscore[t] - mx); sscore[t] = e; sum += e; }
  red[tid] = sum; __syncthreads();
  for (int o = 128; o > 0; o >>= 1) { if (tid < o) red[tid] += red[tid + o]; __syncthreads(); }
  float inv = 1.f / red[0]; __syncthreads();
  for (int t = tid; t < S; t += 256) sscore[t] *= inv;
  __syncthreads();
  for (int cc = tid; cc < HDIM; cc += 256) {
    float a = 0.f;
    for (int t = 0; t < S; ++t) a += sscore[t] * seq[(long)t * HDIM + cc];
    emb[(long)n * HDIM + cc] = a;
  }
}

// ---------------------------------------------------------------- small fp32 GEMM (head MLPs, M=32)
__global__ void k_mm_small(const float* __restrict__ A, const float* __restrict__ W,
                           const float* __restrict__ bias, float* __restrict__ C,
                           int Mr, int Nc, int Kd, int act) {
  int idx = blockIdx.x * blockDim.x + threadIdx.x;
  if (idx >= Mr * Nc) return;
  int m = idx / Nc, n = idx % Nc;
  float s = bias[n];
  const float* ar = A + (long)m * Kd;
  for (int kk = 0; kk < Kd; ++kk) s += ar[kk] * W[(long)kk * Nc + n];
  if (act) s = gelu_exact(s);
  C[idx] = s;
}

// ---------------------------------------------------------------- L2 normalize rows (P=256)
__global__ __launch_bounds__(256) void k_l2norm(const float* __restrict__ proj, float* __restrict__ out) {
  __shared__ float red[256];
  int n = blockIdx.x, tid = threadIdx.x;
  float v = proj[n * PDIM + tid];
  red[tid] = v * v; __syncthreads();
  for (int o = 128; o > 0; o >>= 1) { if (tid < o) red[tid] += red[tid + o]; __syncthreads(); }
  float norm = fmaxf(sqrtf(red[0]), 1e-12f);
  out[n * PDIM + tid] = v / norm;
}

// ================================================================ host
extern "C" void kernel_launch(void* const* d_in, const int* in_sizes, int n_in,
                              void* d_out, int out_size, void* d_ws, size_t ws_size,
                              hipStream_t stream) {
  const int   B  = 2, SD = 2048, SS = 1024;
  const int* doc_ids   = (const int*)d_in[0];
  const int* doc_am    = (const int*)d_in[1];
  const int* sum_ids   = (const int*)d_in[2];
  const int* sum_am    = (const int*)d_in[3];
  const int* og_masks  = (const int*)d_in[4];
  const int* llm_masks = (const int*)d_in[5];
  const int* sidx      = (const int*)d_in[6];
  const float* wemb  = (const float*)d_in[7];
  const float* pemb  = (const float*)d_in[8];
  const float* emb_g = (const float*)d_in[9];
  const float* emb_b = (const float*)d_in[10];
  const float* Wq = (const float*)d_in[11]; const float* bq = (const float*)d_in[12];
  const float* Wk = (const float*)d_in[13]; const float* bk = (const float*)d_in[14];
  const float* Wv = (const float*)d_in[15]; const float* bv = (const float*)d_in[16];
  const float* Wo = (const float*)d_in[17]; const float* bo = (const float*)d_in[18];
  const float* lnag = (const float*)d_in[19]; const float* lnab = (const float*)d_in[20];
  const float* Wf1 = (const float*)d_in[21]; const float* bf1 = (const float*)d_in[22];
  const float* Wf2 = (const float*)d_in[23]; const float* bf2 = (const float*)d_in[24];
  const float* lnfg = (const float*)d_in[25]; const float* lnfb = (const float*)d_in[26];
  const float* pool_w = (const float*)d_in[27];
  const float* pj_g1 = (const float*)d_in[28]; const float* pj_b1 = (const float*)d_in[29];
  const float* pj_W1 = (const float*)d_in[30]; const float* pj_c1 = (const float*)d_in[31];
  const float* pj_g2 = (const float*)d_in[32]; const float* pj_b2 = (const float*)d_in[33];
  const float* pj_W2 = (const float*)d_in[34]; const float* pj_c2 = (const float*)d_in[35];

  char* ws = (char*)d_ws;
  size_t off = 0;
  auto alloc = [&](size_t bytes) { size_t p = off; off += (bytes + 255) & ~(size_t)255; return p; };

  const long HH = (long)HDIM * HDIM;
  const long HF = (long)HDIM * FFDIM;
  const long MmaxH  = (long)B * SD * HDIM;
  const long MmaxFF = (long)B * SD * FFDIM;

  __bf16* WqBf  = (__bf16*)(ws + alloc(NLAYER * HH * 2));
  __bf16* WkBf  = (__bf16*)(ws + alloc(NLAYER * HH * 2));
  __bf16* WvBf  = (__bf16*)(ws + alloc(NLAYER * HH * 2));
  __bf16* WoBf  = (__bf16*)(ws + alloc(NLAYER * HH * 2));
  __bf16* Wf1Bf = (__bf16*)(ws + alloc(NLAYER * HF * 2));
  __bf16* Wf2Bf = (__bf16*)(ws + alloc(NLAYER * HF * 2));
  float*  xdoc  = (float*)(ws + alloc((long)B * SD * HDIM * 4));
  float*  xsum  = (float*)(ws + alloc((long)B * SS * HDIM * 4));
  __bf16* xbf   = (__bf16*)(ws + alloc(MmaxH * 2));
  float*  bigA  = (float*)(ws + alloc(MmaxFF * 4)); // q|k|v|a during attention, h during FFN
  __bf16* abf   = (__bf16*)(ws + alloc(MmaxH * 2));
  float*  t1    = (float*)(ws + alloc(MmaxH * 4));
  __bf16* hbf   = (__bf16*)(ws + alloc(MmaxFF * 2));
  float*  embB  = (float*)(ws + alloc((long)NSPAN * HDIM * 4));
  float*  hhB   = (float*)(ws + alloc((long)NSPAN * HDIM * 4));
  float*  h2B   = (float*)(ws + alloc((long)NSPAN * HDIM * 4));
  float*  projB = (float*)(ws + alloc((long)NSPAN * PDIM * 4));

  float* qb  = bigA;
  float* kb2 = bigA + MmaxH;
  float* vb2 = bigA + 2 * MmaxH;
  float* ab  = bigA + 3 * MmaxH;
  float* hb  = bigA;                       // reuse after attention done

  // one-shot: convert + pack weights into WMMA fragment layout (~85MB bf16, L2-resident)
  k_pack_w<<<2048, 256, 0, stream>>>(Wq,  WqBf,  NLAYER, HDIM,  HDIM);
  k_pack_w<<<2048, 256, 0, stream>>>(Wk,  WkBf,  NLAYER, HDIM,  HDIM);
  k_pack_w<<<2048, 256, 0, stream>>>(Wv,  WvBf,  NLAYER, HDIM,  HDIM);
  k_pack_w<<<2048, 256, 0, stream>>>(Wo,  WoBf,  NLAYER, HDIM,  HDIM);
  k_pack_w<<<2048, 256, 0, stream>>>(Wf1, Wf1Bf, NLAYER, HDIM,  FFDIM);
  k_pack_w<<<2048, 256, 0, stream>>>(Wf2, Wf2Bf, NLAYER, FFDIM, HDIM);

  auto encode = [&](const int* ids, const int* am, int S, float* x) {
    int Mrows = B * S;
    k_embed<<<Mrows, 256, 0, stream>>>(ids, wemb, pemb, x, S);
    k_ln<<<Mrows, 256, 0, stream>>>(x, nullptr, emb_g, emb_b, x, xbf);
    int nc = S / WINSZ;
    dim3 gH(HDIM / 64, Mrows / 128);
    dim3 gF(FFDIM / 64, Mrows / 128);
    dim3 gA(16, nc, B * NHEAD);
    for (int l = 0; l < NLAYER; ++l) {
      const __bf16* wq = WqBf  + (long)l * HH;
      const __bf16* wk = WkBf  + (long)l * HH;
      const __bf16* wv = WvBf  + (long)l * HH;
      const __bf16* wo = WoBf  + (long)l * HH;
      const __bf16* w1 = Wf1Bf + (long)l * HF;
      const __bf16* w2 = Wf2Bf + (long)l * HF;
      k_gemm_bf16<<<gH, 256, 0, stream>>>(xbf, wq, bq + l * HDIM, qb,  nullptr, Mrows, HDIM, HDIM, 0);
      k_gemm_bf16<<<gH, 256, 0, stream>>>(xbf, wk, bk + l * HDIM, kb2, nullptr, Mrows, HDIM, HDIM, 0);
      k_gemm_bf16<<<gH, 256, 0, stream>>>(xbf, wv, bv + l * HDIM, vb2, nullptr, Mrows, HDIM, HDIM, 0);
      k_attn<<<gA, 32, 0, stream>>>(qb, kb2, vb2, am, ab, abf, S);
      k_gemm_bf16<<<gH, 256, 0, stream>>>(abf, wo, bo + l * HDIM, t1, nullptr, Mrows, HDIM, HDIM, 0);
      k_ln<<<Mrows, 256, 0, stream>>>(x, t1, lnag + l * HDIM, lnab + l * HDIM, x, xbf);
      k_gemm_bf16<<<gF, 256, 0, stream>>>(xbf, w1, bf1 + l * FFDIM, hb, hbf, Mrows, FFDIM, HDIM, 1);
      k_gemm_bf16<<<gH, 256, 0, stream>>>(hbf, w2, bf2 + l * HDIM, t1, nullptr, Mrows, HDIM, FFDIM, 0);
      k_ln<<<Mrows, 256, 0, stream>>>(x, t1, lnfg + l * HDIM, lnfb + l * HDIM, x, xbf);
    }
  };

  encode(doc_ids, doc_am, SD, xdoc);
  encode(sum_ids, sum_am, SS, xsum);

  auto heads = [&](const float* enc, const int* smask, int S, float* outp) {
    k_span_pool<<<NSPAN, 256, 0, stream>>>(enc, smask, sidx, pool_w, embB, S);
    k_ln<<<NSPAN, 256, 0, stream>>>(embB, nullptr, pj_g1, pj_b1, hhB, nullptr);
    k_mm_small<<<(NSPAN * HDIM + 255) / 256, 256, 0, stream>>>(hhB, pj_W1, pj_c1, h2B, NSPAN, HDIM, HDIM, 1);
    k_ln<<<NSPAN, 256, 0, stream>>>(h2B, nullptr, pj_g2, pj_b2, hhB, nullptr);
    k_mm_small<<<(NSPAN * PDIM + 255) / 256, 256, 0, stream>>>(hhB, pj_W2, pj_c2, projB, NSPAN, PDIM, HDIM, 0);
    k_l2norm<<<NSPAN, 256, 0, stream>>>(projB, outp);
  };

  float* outF = (float*)d_out;
  heads(xdoc, og_masks,  SD, outF);
  heads(xsum, llm_masks, SS, outF + NSPAN * PDIM);
}
